// LayerWithLoRAMixture_57483842289993
// MI455X (gfx1250) — matrix operand order repaired and checked by reading, not compiled, not run on hardware
//
#include <hip/hip_runtime.h>

// MI455X (gfx1250) LoRA-mixture layer.
//   out = x@W^T + b + (x@A_cat^T) @ Bp^T,  Bp folded with router probs.
// HBM-bound (~600MB traffic vs ~2.2 TFLOP): use bf16x3 split WMMA
// (v_wmma_f32_16x16x32_bf16) for ~fp32 accuracy at negligible extra cost.

typedef __attribute__((ext_vector_type(16))) __bf16 v16bf;
typedef __attribute__((ext_vector_type(8)))  float  v8f;

#define M_TOT  16384   // B_SZ * S
#define N_MAIN 4096    // D_OUT
#define K_MAIN 4096    // D_IN
#define ER     256     // E * R
#define BM     128
#define BN     128
#define BK     32
#define LDSS   36      // LDS row stride in halves (32 + 4 pad: 8B-aligned rows, conflict-free frags)

__device__ __forceinline__ unsigned short f2bf(float f) {
  union { float f; unsigned u; } c; c.f = f;
  unsigned u = c.u + 0x7FFFu + ((c.u >> 16) & 1u);   // round-to-nearest-even
  return (unsigned short)(u >> 16);
}
__device__ __forceinline__ float bf2f(unsigned short h) {
  union { float f; unsigned u; } c; c.u = ((unsigned)h) << 16; return c.f;
}

// Split one float4 into bf16 hi/lo and store 4 packed halves to LDS.
__device__ __forceinline__ void storeSplit4(unsigned short* hiB, unsigned short* loB,
                                            int row, int c4, float4 x) {
  float xs[4] = {x.x, x.y, x.z, x.w};
  unsigned short h[4], l[4];
#pragma unroll
  for (int i = 0; i < 4; ++i) {
    h[i] = f2bf(xs[i]);
    l[i] = f2bf(xs[i] - bf2f(h[i]));
  }
  uint2 hp, lp;
  hp.x = h[0] | ((unsigned)h[1] << 16); hp.y = h[2] | ((unsigned)h[3] << 16);
  lp.x = l[0] | ((unsigned)l[1] << 16); lp.y = l[2] | ((unsigned)l[3] << 16);
  *(uint2*)(hiB + row * LDSS + c4 * 4) = hp;
  *(uint2*)(loB + row * LDSS + c4 * 4) = lp;
}

// A-fragment (16x32 bf16), ISA 7.12.2: lane L holds row M=L%16;
// VGPR v: K = (v>>2)*16 + (lane>=16?8:0) + (v&3)*2, two halves packed.
__device__ __forceinline__ v16bf load_fragA(const unsigned short* base, int row0, int lane) {
  union { v16bf v; unsigned u[8]; } fr;
  const int half = lane >> 4;
  const unsigned short* p = base + (row0 + (lane & 15)) * LDSS + (half << 3);
#pragma unroll
  for (int v = 0; v < 8; ++v) {
    int kk = ((v >> 2) << 4) + ((v & 3) << 1);
    fr.u[v] = *(const unsigned*)(p + kk);
  }
  return fr.v;
}

// B-fragment (32x16 bf16): lane L holds column N=L%16 (= a row of W);
// VGPR v: K = (v>>2)*16 + (lane>=16?8:0) + (v&3)*2, two halves packed (same as A).
__device__ __forceinline__ v16bf load_fragB(const unsigned short* base, int row0, int lane) {
  union { v16bf v; unsigned u[8]; } fr;
  const int half = lane >> 4;
  const unsigned short* p = base + (row0 + (lane & 15)) * LDSS + (half << 3);
#pragma unroll
  for (int v = 0; v < 8; ++v)
    fr.u[v] = *(const unsigned*)(p + ((v >> 2) << 4) + ((v & 3) << 1));
  return fr.v;
}

#define WMMA_BF16(A, B, C) \
  __builtin_amdgcn_wmma_f32_16x16x32_bf16(false, (A), false, (B), (short)0, (C), false, false)

// Fold router probs into B: Bp[o][e*16+r] = probs[e] * B[e][o][r], bf16 hi/lo.
__global__ __launch_bounds__(256) void fold_bp_kernel(
    const float* __restrict__ Bm, const float* __restrict__ probs,
    unsigned short* __restrict__ bp_hi, unsigned short* __restrict__ bp_lo) {
  int idx = blockIdx.x * 256 + threadIdx.x;       // idx = o*256 + j
  int o = idx >> 8, j = idx & 255, e = j >> 4, r = j & 15;
  float v = probs[e] * Bm[((size_t)e * N_MAIN + o) * 16 + r];
  unsigned short hi = f2bf(v);
  bp_hi[idx] = hi;
  bp_lo[idx] = f2bf(v - bf2f(hi));
}

// bf16x3 split GEMM: D = Aop[M,K] * Bop[N,K]^T (both row-major, fp32).
// MODE 0: write bf16 hi/lo pair (producing h).
// MODE 1: accumulate the rank-ER tail (h @ Bp^T) from pre-split bf16 workspace,
//         add bias, write fp32 out.
template <int MODE>
__global__ __launch_bounds__(256) void gemm_bf16x3_kernel(
    const float* __restrict__ Aop, const float* __restrict__ Bop,
    int N, int K, int ldA, int ldB,
    const unsigned short* __restrict__ h_hi, const unsigned short* __restrict__ h_lo,
    const unsigned short* __restrict__ bp_hi, const unsigned short* __restrict__ bp_lo,
    const float* __restrict__ bias,
    float* __restrict__ outF,
    unsigned short* __restrict__ out_hi, unsigned short* __restrict__ out_lo) {
  __shared__ unsigned short sA_hi[BM * LDSS];
  __shared__ unsigned short sA_lo[BM * LDSS];
  __shared__ unsigned short sB_hi[BN * LDSS];
  __shared__ unsigned short sB_lo[BN * LDSS];

  const int t    = threadIdx.x;
  const int lane = t & 31;
  const int wave = t >> 5;          // 8 waves: 4 along M x 2 along N
  const int wm   = wave & 3;        // 32 rows each
  const int wn   = wave >> 2;       // 64 cols each
  const int m0   = blockIdx.y * BM;
  const int n0   = blockIdx.x * BN;

  v8f acc[2][4];
#pragma unroll
  for (int i = 0; i < 2; ++i)
#pragma unroll
    for (int j = 0; j < 4; ++j) acc[i][j] = (v8f)(0.0f);

  // ---------------- main K loop: fp32 operands, split on the fly -----------
  for (int k0 = 0; k0 < K; k0 += BK) {
    __syncthreads();
#pragma unroll
    for (int i = 0; i < 4; ++i) {
      int idx = t + i * 256;        // 1024 float4 groups per operand tile
      int row = idx >> 3;
      int c4  = idx & 7;
      float4 xa = *(const float4*)(Aop + (size_t)(m0 + row) * ldA + k0 + c4 * 4);
      storeSplit4(sA_hi, sA_lo, row, c4, xa);
      float4 xb = *(const float4*)(Bop + (size_t)(n0 + row) * ldB + k0 + c4 * 4);
      storeSplit4(sB_hi, sB_lo, row, c4, xb);
    }
    __syncthreads();

    v16bf ah0 = load_fragA(sA_hi, wm * 32 +  0, lane);
    v16bf ah1 = load_fragA(sA_hi, wm * 32 + 16, lane);
    v16bf al0 = load_fragA(sA_lo, wm * 32 +  0, lane);
    v16bf al1 = load_fragA(sA_lo, wm * 32 + 16, lane);
#pragma unroll
    for (int j = 0; j < 4; ++j) {
      v16bf bh = load_fragB(sB_hi, wn * 64 + j * 16, lane);
      v16bf bl = load_fragB(sB_lo, wn * 64 + j * 16, lane);
      acc[0][j] = WMMA_BF16(ah0, bh, acc[0][j]);
      acc[0][j] = WMMA_BF16(ah0, bl, acc[0][j]);
      acc[0][j] = WMMA_BF16(al0, bh, acc[0][j]);
      acc[1][j] = WMMA_BF16(ah1, bh, acc[1][j]);
      acc[1][j] = WMMA_BF16(ah1, bl, acc[1][j]);
      acc[1][j] = WMMA_BF16(al1, bh, acc[1][j]);
    }
  }

  // ---------------- rank-ER tail: pre-split bf16 h / Bp --------------------
  if constexpr (MODE == 1) {
    for (int k0 = 0; k0 < ER; k0 += BK) {
      __syncthreads();
#pragma unroll
      for (int i = 0; i < 4; ++i) {
        int idx = t + i * 256;
        int row = idx >> 3;
        int g   = idx & 7;          // 8 groups of 4 halves per 32-wide row
        size_t ga = (size_t)(m0 + row) * ER + k0 + g * 4;
        size_t gb = (size_t)(n0 + row) * ER + k0 + g * 4;
        *(uint2*)(sA_hi + row * LDSS + g * 4) = *(const uint2*)(h_hi + ga);
        *(uint2*)(sA_lo + row * LDSS + g * 4) = *(const uint2*)(h_lo + ga);
        *(uint2*)(sB_hi + row * LDSS + g * 4) = *(const uint2*)(bp_hi + gb);
        *(uint2*)(sB_lo + row * LDSS + g * 4) = *(const uint2*)(bp_lo + gb);
      }
      __syncthreads();

      v16bf ah0 = load_fragA(sA_hi, wm * 32 +  0, lane);
      v16bf ah1 = load_fragA(sA_hi, wm * 32 + 16, lane);
      v16bf al0 = load_fragA(sA_lo, wm * 32 +  0, lane);
      v16bf al1 = load_fragA(sA_lo, wm * 32 + 16, lane);
#pragma unroll
      for (int j = 0; j < 4; ++j) {
        v16bf bh = load_fragB(sB_hi, wn * 64 + j * 16, lane);
        v16bf bl = load_fragB(sB_lo, wn * 64 + j * 16, lane);
        acc[0][j] = WMMA_BF16(ah0, bh, acc[0][j]);
        acc[0][j] = WMMA_BF16(ah0, bl, acc[0][j]);
        acc[0][j] = WMMA_BF16(al0, bh, acc[0][j]);
        acc[1][j] = WMMA_BF16(ah1, bh, acc[1][j]);
        acc[1][j] = WMMA_BF16(ah1, bl, acc[1][j]);
        acc[1][j] = WMMA_BF16(al1, bh, acc[1][j]);
      }
    }
  }

  // ---------------- epilogue (C/D layout: VGPR v -> M = v or v+8) ----------
#pragma unroll
  for (int i = 0; i < 2; ++i) {
#pragma unroll
    for (int j = 0; j < 4; ++j) {
      int tm  = m0 + wm * 32 + i * 16;
      int col = n0 + wn * 64 + j * 16 + (lane & 15);
      float bv = (MODE == 1) ? bias[col] : 0.0f;
#pragma unroll
      for (int v = 0; v < 8; ++v) {
        int row = tm + ((lane < 16) ? v : v + 8);
        float val = acc[i][j][v];
        if constexpr (MODE == 1) {
          outF[(size_t)row * N + col] = val + bv;
        } else {
          unsigned short hi = f2bf(val);
          out_hi[(size_t)row * N + col] = hi;
          out_lo[(size_t)row * N + col] = f2bf(val - bf2f(hi));
        }
      }
    }
  }
}

extern "C" void kernel_launch(void* const* d_in, const int* in_sizes, int n_in,
                              void* d_out, int out_size, void* d_ws, size_t ws_size,
                              hipStream_t stream) {
  (void)in_sizes; (void)n_in; (void)out_size; (void)ws_size;
  const float* x     = (const float*)d_in[0];   // [16384, 4096]
  const float* W     = (const float*)d_in[1];   // [4096, 4096]
  const float* bias  = (const float*)d_in[2];   // [4096]
  const float* Acat  = (const float*)d_in[3];   // [E,R,D] == [256, 4096]
  const float* Bm    = (const float*)d_in[4];   // [E, 4096, 16]
  const float* probs = (const float*)d_in[5];   // [16]
  float* out = (float*)d_out;

  // workspace: h hi/lo [M,256] bf16, Bp hi/lo [N,256] bf16  (~21 MB)
  unsigned short* h_hi  = (unsigned short*)d_ws;
  unsigned short* h_lo  = h_hi  + (size_t)M_TOT * ER;
  unsigned short* bp_hi = h_lo  + (size_t)M_TOT * ER;
  unsigned short* bp_lo = bp_hi + (size_t)N_MAIN * ER;

  // 1) fold probs into B
  fold_bp_kernel<<<(N_MAIN * ER) / 256, 256, 0, stream>>>(Bm, probs, bp_hi, bp_lo);

  // 2) h = x @ A_cat^T  (M x 256, K=4096), stored as bf16 hi/lo
  gemm_bf16x3_kernel<0><<<dim3(ER / BN, M_TOT / BM), 256, 0, stream>>>(
      x, Acat, ER, K_MAIN, K_MAIN, K_MAIN,
      nullptr, nullptr, nullptr, nullptr, nullptr,
      nullptr, h_hi, h_lo);

  // 3) out = x @ W^T + h @ Bp^T + bias
  gemm_bf16x3_kernel<1><<<dim3(N_MAIN / BN, M_TOT / BM), 256, 0, stream>>>(
      x, W, N_MAIN, K_MAIN, K_MAIN, K_MAIN,
      h_hi, h_lo, bp_hi, bp_lo, bias,
      out, nullptr, nullptr);
}
